// VectorQuantizerEMA_36429912605194
// MI455X (gfx1250) — matrix operand order repaired and checked by reading, not compiled, not run on hardware
//
#include <hip/hip_runtime.h>

#define EMB_D   256
#define NUM_K   1024
#define NTOK    65536        // 16*64*64
#define DECAYF  0.99f
#define EPSF    1e-5f

typedef __attribute__((ext_vector_type(16))) _Float16     v16h;
typedef __attribute__((ext_vector_type(8)))  _Float16     v8h;
typedef __attribute__((ext_vector_type(8)))  float        v8f;
typedef __attribute__((ext_vector_type(4)))  float        f4;
typedef __attribute__((ext_vector_type(4)))  unsigned int v4u;
typedef __attribute__((ext_vector_type(8)))  int          v8i_t;
typedef __attribute__((ext_vector_type(4)))  int          v4i_t;

// ---------------------------------------------------------------------------
// TDM: DMA one contiguous 8 KB block (16 codes x 256 dims, f16) into LDS.
// D# per CDNA5 ISA:
//  group0: w0 = count(1:0)=1 ; w1 = lds_addr ; w2 = gaddr[31:0] ;
//          w3 = gaddr[56:32] | type(127:126)=2
//  group1: w0 = data_size(17:16)=1 (2B) ; tensor_dim0(79:48)=1<<18 ->
//          w1[31:16]=0, w2[15:0]=4 ; tensor_dim1(111:80)=1 -> w2[31:16]=1 ;
//          tile_dim0(127:112)=4096 -> w3[31:16] ; tile_dim1(143:128)=1 ->
//          w4[15:0] ; tensor_dim0_stride(207:160)=1<<18 -> w5
//  groups 2/3: zero (<=2D tensor)
// ---------------------------------------------------------------------------
__device__ __forceinline__ void tdm_load_tile8k(const _Float16* gsrc, void* ldst)
{
    const unsigned long long ga = (unsigned long long)(uintptr_t)gsrc;
    const unsigned int       la = (unsigned int)(uintptr_t)ldst;

    v4u g0 = { 1u, la, (unsigned int)ga,
               (unsigned int)((ga >> 32) & 0x01FFFFFFu) | (2u << 30) };
    v8i_t g1 = { 0x00010000, 0, (int)((1u << 16) | 4u), (int)(4096u << 16),
                 1, (int)(1u << 18), 0, 0 };
    v4i_t z4 = { 0, 0, 0, 0 };
#if __clang_major__ >= 23
    v8i_t z8 = { 0, 0, 0, 0, 0, 0, 0, 0 };
    __builtin_amdgcn_tensor_load_to_lds(g0, g1, z4, z4, z8, 0);
#else
    __builtin_amdgcn_tensor_load_to_lds(g0, g1, z4, z4, 0);
#endif
}

// ---------------------------------------------------------------------------
// Kernel 0: codebook -> f16, per-code squared norms, zero scatter accumulators
// ---------------------------------------------------------------------------
__global__ __launch_bounds__(EMB_D) void vq_prep(
    const float* __restrict__ codebook,
    _Float16* __restrict__ cb_h,
    float* __restrict__ cnorm,
    float* __restrict__ acc_w,
    float* __restrict__ acc_c)
{
    const int k = blockIdx.x;
    const int d = threadIdx.x;
    const float v = codebook[(size_t)k * EMB_D + d];
    cb_h[(size_t)k * EMB_D + d]  = (_Float16)v;
    acc_w[(size_t)k * EMB_D + d] = 0.0f;

    __shared__ float red[EMB_D];
    red[d] = v * v;
    __syncthreads();
    for (int s = EMB_D / 2; s > 0; s >>= 1) {
        if (d < s) red[d] += red[d + s];
        __syncthreads();
    }
    if (d == 0) { cnorm[k] = red[0]; acc_c[k] = 0.0f; }
}

// ---------------------------------------------------------------------------
// Kernel 1: main — WMMA distances + argmin + one-hot + gather + scatter-add
// Block = 128 threads (4 waves). Each wave owns 16 tokens; block owns 64.
// Codebook tiles arrive via double-buffered TDM (tensor_load_to_lds).
// ---------------------------------------------------------------------------
__global__ __launch_bounds__(128) void vq_main(
    const float*    __restrict__ x,
    const float*    __restrict__ codebook,
    const _Float16* __restrict__ cb_h,
    const float*    __restrict__ cnorm,
    float* __restrict__ out_discrete,
    float* __restrict__ out_quant,
    float* __restrict__ acc_w,
    float* __restrict__ acc_c)
{
    __shared__ _Float16 Xs[4][16][EMB_D];     // 32 KB: 64 tokens, f16
    __shared__ _Float16 Bs[2][16][EMB_D];     // 16 KB: double-buffered B tiles
    __shared__ int      sIdx[64];

    const int tid  = threadIdx.x;
    const int w    = tid >> 5;      // wave in block
    const int lane = tid & 31;
    const int hl   = lane >> 4;     // lane half (0/1)
    const int n    = lane & 15;     // column / row-within-half index
    const int tokBase = blockIdx.x * 64;

    // ---- kick off TDM for B tile 0 (overlaps with x staging below) --------
    if (w == 0) tdm_load_tile8k(cb_h, &Bs[0][0][0]);

    // ---- stage 64 tokens of x into LDS as f16 -----------------------------
    {
        const f4* xg = (const f4*)(x + (size_t)tokBase * EMB_D);
        _Float16* xs = &Xs[0][0][0];
        for (int i = tid; i < 64 * (EMB_D / 4); i += 128) {
            f4 f = xg[i];
            _Float16* p = xs + i * 4;
            p[0] = (_Float16)f.x; p[1] = (_Float16)f.y;
            p[2] = (_Float16)f.z; p[3] = (_Float16)f.w;
        }
    }
    if (w == 0) __builtin_amdgcn_s_wait_tensorcnt(0);
    __syncthreads();

    // ---- A fragments (16x32 f16 per chunk), resident in VGPRs -------------
    // A layout: lane holds row M = lane%16; elem j -> K = hl*16 + j.
    //           Two 16-byte LDS reads each.
    v16h a[8];
    #pragma unroll
    for (int ch = 0; ch < 8; ++ch) {
        const v8h lo = *(const v8h*)&Xs[w][n][ch * 32 + hl * 16];
        const v8h hi = *(const v8h*)&Xs[w][n][ch * 32 + hl * 16 + 8];
        a[ch] = __builtin_shufflevector(lo, hi,
                 0,1,2,3,4,5,6,7,8,9,10,11,12,13,14,15);
    }

    float bestVal[8];
    int   bestCode[8];
    #pragma unroll
    for (int r = 0; r < 8; ++r) { bestVal[r] = 3.4e38f; bestCode[r] = 0; }

    // ---- sweep codebook in 16-code tiles, double-buffered TDM -------------
    for (int t = 0; t < NUM_K / 16; ++t) {
        const int c0  = t * 16;
        const int cur = t & 1;

        // prefetch next tile while this one computes
        if (w == 0 && t + 1 < NUM_K / 16)
            tdm_load_tile8k(cb_h + (size_t)(c0 + 16) * EMB_D,
                            &Bs[cur ^ 1][0][0]);

        // B layout: lane = column N (lane%16), K = hl*16 + j -> one
        // contiguous 32-byte LDS read. Bs[cur][n][k] = B(k, n) (codebook^T).
        v16h bfr[8];
        #pragma unroll
        for (int ch = 0; ch < 8; ++ch)
            bfr[ch] = *(const v16h*)&Bs[cur][n][ch * 32 + hl * 16];

        v8f acc = {};
        #pragma unroll
        for (int ch = 0; ch < 8; ++ch)
            acc = __builtin_amdgcn_wmma_f32_16x16x32_f16(
                      false, a[ch], false, bfr[ch], (short)0, acc, false, false);

        const float cn = cnorm[c0 + n];
        #pragma unroll
        for (int r = 0; r < 8; ++r) {
            // argmin over ||x||^2 - 2 x.c + ||c||^2 == argmin over cn - 2 dot
            const float score = cn - 2.0f * acc[r];
            if (score < bestVal[r]) { bestVal[r] = score; bestCode[r] = c0 + n; }
        }

        // wave0 folds the DMA-complete wait into its barrier arrival
        if (w == 0) __builtin_amdgcn_s_wait_tensorcnt(0);
        __syncthreads();
    }

    // ---- cross-lane argmin within each half (16 columns) ------------------
    // C/D layout: VGPR r -> token row M = hl*8 + r, column N = lane%16.
    #pragma unroll
    for (int r = 0; r < 8; ++r) {
        float v = bestVal[r];
        int   c = bestCode[r];
        for (int off = 8; off > 0; off >>= 1) {
            const float ov = __shfl_xor(v, off, 16);
            const int   oc = __shfl_xor(c, off, 16);
            if (ov < v || (ov == v && oc < c)) { v = ov; c = oc; }
        }
        if (n == 0) sIdx[w * 16 + hl * 8 + r] = c;
    }
    __syncthreads();

    // ---- emit one-hot rows (256 MB total -> nontemporal) ------------------
    for (int i = tid; i < 64 * (NUM_K / 4); i += 128) {
        const int row = i >> 8;           // NUM_K/4 = 256 per row
        const int c4  = i & 255;
        const int idx = sIdx[row];
        f4 val = {0.0f, 0.0f, 0.0f, 0.0f};
        const int base = c4 * 4;
        if (idx >= base && idx < base + 4) val[idx - base] = 1.0f;
        f4* dst = (f4*)(out_discrete + (size_t)(tokBase + row) * NUM_K) + c4;
        __builtin_nontemporal_store(val, dst);
    }

    // ---- quantized gather + scatter-add of x into code accumulators -------
    for (int i = tid; i < 64 * (EMB_D / 4); i += 128) {
        const int row = i >> 6;           // EMB_D/4 = 64 per row
        const int d4  = i & 63;
        const int idx = sIdx[row];
        const f4 q = *((const f4*)(codebook + (size_t)idx * EMB_D) + d4);
        __builtin_nontemporal_store(
            q, (f4*)(out_quant + (size_t)(tokBase + row) * EMB_D) + d4);

        const f4 xv = *((const f4*)(x + (size_t)(tokBase + row) * EMB_D) + d4);
        float* aw = acc_w + (size_t)idx * EMB_D + d4 * 4;
        atomicAdd(aw + 0, xv.x); atomicAdd(aw + 1, xv.y);
        atomicAdd(aw + 2, xv.z); atomicAdd(aw + 3, xv.w);
    }

    if (tid < 64) atomicAdd(&acc_c[sIdx[tid]], 1.0f);
}

// ---------------------------------------------------------------------------
// Kernel 2: EMA count + Laplace normalization (single block reduction)
// ---------------------------------------------------------------------------
__global__ __launch_bounds__(NUM_K) void vq_fin_count(
    const float* __restrict__ ema_count,
    const float* __restrict__ acc_c,
    float* __restrict__ out_count)
{
    const int k = threadIdx.x;
    const float nc = ema_count[k] * DECAYF + acc_c[k] * (1.0f - DECAYF);

    __shared__ float red[NUM_K];
    red[k] = nc;
    __syncthreads();
    for (int s = NUM_K / 2; s > 0; s >>= 1) {
        if (k < s) red[k] += red[k + s];
        __syncthreads();
    }
    const float nsum = red[0];
    out_count[k] = (nc + EPSF) / (nsum + NUM_K * EPSF) * nsum;
}

// ---------------------------------------------------------------------------
// Kernel 3: EMA weight + normalized codebook
// ---------------------------------------------------------------------------
__global__ __launch_bounds__(EMB_D) void vq_fin_weight(
    const float* __restrict__ ema_weight,
    const float* __restrict__ acc_w,
    const float* __restrict__ out_count,
    float* __restrict__ out_weight,
    float* __restrict__ out_codebook)
{
    const int k = blockIdx.x;
    const int d = threadIdx.x;
    const size_t i = (size_t)k * EMB_D + d;
    const float nw = ema_weight[i] * DECAYF + acc_w[i] * (1.0f - DECAYF);
    out_weight[i]   = nw;
    out_codebook[i] = nw / out_count[k];
}

// ---------------------------------------------------------------------------
extern "C" void kernel_launch(void* const* d_in, const int* in_sizes, int n_in,
                              void* d_out, int out_size, void* d_ws, size_t ws_size,
                              hipStream_t stream)
{
    (void)in_sizes; (void)n_in; (void)out_size; (void)ws_size;

    const float* x          = (const float*)d_in[0];
    const float* codebook   = (const float*)d_in[1];
    const float* ema_count  = (const float*)d_in[2];
    const float* ema_weight = (const float*)d_in[3];

    float* out          = (float*)d_out;
    float* out_discrete = out;                                   // 65536*1024
    float* out_quant    = out_discrete + (size_t)NTOK * NUM_K;   // 65536*256
    float* out_count    = out_quant    + (size_t)NTOK * EMB_D;   // 1024
    float* out_weight   = out_count    + NUM_K;                  // 1024*256
    float* out_codebook = out_weight   + (size_t)NUM_K * EMB_D;  // 1024*256

    char* ws = (char*)d_ws;
    float* cnorm = (float*)ws;  ws += (size_t)NUM_K * sizeof(float);
    float* acc_c = (float*)ws;  ws += (size_t)NUM_K * sizeof(float);
    float* acc_w = (float*)ws;  ws += (size_t)NUM_K * EMB_D * sizeof(float);
    _Float16* cb_h = (_Float16*)ws;   // NUM_K * EMB_D f16

    vq_prep<<<NUM_K, EMB_D, 0, stream>>>(codebook, cb_h, cnorm, acc_w, acc_c);
    vq_main<<<NTOK / 64, 128, 0, stream>>>(x, codebook, cb_h, cnorm,
                                           out_discrete, out_quant, acc_w, acc_c);
    vq_fin_count<<<1, NUM_K, 0, stream>>>(ema_count, acc_c, out_count);
    vq_fin_weight<<<NUM_K, EMB_D, 0, stream>>>(ema_weight, acc_w, out_count,
                                               out_weight, out_codebook);
}
